// GAT_55697135894617
// MI455X (gfx1250) — compile-verified, matrix operand order checked by reading, not run on hardware
//
#include <hip/hip_runtime.h>

typedef _Float16 v16h __attribute__((ext_vector_type(16)));
typedef _Float16 v8h  __attribute__((ext_vector_type(8)));
typedef float    v8f  __attribute__((ext_vector_type(8)));

#define NN 50000
#define NE 800000
#define HC 128
#define NH 4
#define CC 32
#define EDIM 8
#define BN_EPS 1e-5f

// ---------------------------------------------------------------- utilities

__global__ void zero_f32_kernel(float* __restrict__ p, int n) {
  int t = blockIdx.x * blockDim.x + threadIdx.x;
  if (t < n) p[t] = 0.0f;
}

__global__ void f32_to_f16_kernel(const float* __restrict__ x,
                                  _Float16* __restrict__ y, int n) {
  int t = blockIdx.x * blockDim.x + threadIdx.x;
  if (t < n) y[t] = (_Float16)x[t];
}

// Pack f32 weight matrix W[K=128][ncolt*16] into WMMA B-operand tiles:
// flat = ((ct*4 + ks)*32 + lane)*16 + j
// lane holds column n = ct*16 + (lane&15); g = lane>>4
// k = ks*32 + (j<8 ? g*8+j : 16 + g*8 + (j-8))
__global__ void pack_w_kernel(const float* __restrict__ w,
                              _Float16* __restrict__ wp, int ncolt) {
  int t = blockIdx.x * blockDim.x + threadIdx.x;
  int total = ncolt * 4 * 32 * 16;
  if (t >= total) return;
  int j    = t & 15;
  int lane = (t >> 4) & 31;
  int ks   = (t >> 9) & 3;
  int ct   = t >> 11;
  int g    = lane >> 4;
  int k    = ks * 32 + ((j < 8) ? (g * 8 + j) : (16 + g * 8 + (j - 8)));
  int n    = ct * 16 + (lane & 15);
  wp[t] = (_Float16)w[k * (ncolt * 16) + n];
}

// ---------------------------------------------------------------- WMMA GEMM
// Y[N x (ncolt*16)] = A_f16[N x 128] @ Wpacked + bias
// grid.x = N/16 row tiles, blockDim = 32*ncolt (one wave per column tile)
__global__ void wmma_gemm_kernel(const _Float16* __restrict__ A,
                                 const _Float16* __restrict__ WP,
                                 const float* __restrict__ bias,
                                 float* __restrict__ Y, int ncolt) {
  const int lane  = threadIdx.x & 31;
  const int ct    = threadIdx.x >> 5;
  const int mBase = blockIdx.x * 16;
  const int m     = mBase + (lane & 15);
  const int g     = lane >> 4;
  const int ldY   = ncolt * 16;

  v8f acc = {};
  const _Float16* arow = A + (size_t)m * HC;
#pragma unroll
  for (int ks = 0; ks < 4; ++ks) {
    union { v16h v; v8h h[2]; } a, b;
    const int kbase = ks * 32;
    a.h[0] = *(const v8h*)(arow + kbase + g * 8);
    a.h[1] = *(const v8h*)(arow + kbase + 16 + g * 8);
    b.v = *(const v16h*)(WP + ((size_t)((ct * 4 + ks) * 32 + lane)) * 16);
    acc = __builtin_amdgcn_wmma_f32_16x16x32_f16(false, a.v, false, b.v,
                                                 (short)0, acc, false, false);
  }
  const int   n  = ct * 16 + (lane & 15);
  const float bv = bias ? bias[n] : 0.0f;
  const int   r0 = 8 * g;
#pragma unroll
  for (int r = 0; r < 8; ++r)
    Y[(size_t)(mBase + r0 + r) * ldY + n] = acc[r] + bv;
}

// ---------------------------------------------------------------- BatchNorm

__global__ void bn_stats_kernel(const float* __restrict__ X,
                                float* __restrict__ sum,
                                float* __restrict__ ssq) {
  int ch = threadIdx.x;  // 128 channels
  float s = 0.f, q = 0.f;
  for (int r = blockIdx.x; r < NN; r += gridDim.x) {
    float v = X[(size_t)r * HC + ch];
    s += v;
    q += v * v;
  }
  atomicAdd(&sum[ch], s);
  atomicAdd(&ssq[ch], q);
}

__global__ void bn_finalize_kernel(const float* __restrict__ sum,
                                   const float* __restrict__ ssq,
                                   float* __restrict__ mean,
                                   float* __restrict__ rs) {
  int ch = threadIdx.x;
  float m = sum[ch] * (1.0f / NN);
  float v = ssq[ch] * (1.0f / NN) - m * m;
  mean[ch] = m;
  rs[ch]   = rsqrtf(v + BN_EPS);
}

__global__ void bn_act_f32_kernel(const float* __restrict__ X,
                                  const float* __restrict__ mean,
                                  const float* __restrict__ rs,
                                  const float* __restrict__ g,
                                  const float* __restrict__ b,
                                  float* __restrict__ Y) {
  int t = blockIdx.x * blockDim.x + threadIdx.x;
  if (t >= NN * HC) return;
  int ch = t & (HC - 1);
  float v = g[ch] * (X[t] - mean[ch]) * rs[ch] + b[ch];
  Y[t] = v > 0.f ? v : 0.01f * v;
}

__global__ void bn_act_f16_kernel(const float* __restrict__ X,
                                  const float* __restrict__ mean,
                                  const float* __restrict__ rs,
                                  const float* __restrict__ g,
                                  const float* __restrict__ b,
                                  _Float16* __restrict__ Y) {
  int t = blockIdx.x * blockDim.x + threadIdx.x;
  if (t >= NN * HC) return;
  int ch = t & (HC - 1);
  float v = g[ch] * (X[t] - mean[ch]) * rs[ch] + b[ch];
  v = v > 0.f ? v : 0.01f * v;
  Y[t] = (_Float16)v;
}

// ---------------------------------------------------------------- edge phase

__device__ __forceinline__ unsigned mapf(float f) {
  unsigned u = __float_as_uint(f);
  return (u & 0x80000000u) ? ~u : (u | 0x80000000u);
}
__device__ __forceinline__ float unmapf(unsigned u) {
  return __uint_as_float((u & 0x80000000u) ? (u ^ 0x80000000u) : ~u);
}

// thread per (edge, head): logit + atomic segment-max over dst
__global__ void edge_logits_kernel(const int* __restrict__ ei,
                                   const float* __restrict__ XL,
                                   const float* __restrict__ XR,
                                   const float* __restrict__ EW,
                                   const float* __restrict__ We,   // [8][128]
                                   const float* __restrict__ att,  // [128]
                                   float* __restrict__ LG,
                                   unsigned* __restrict__ MX) {
  int t = blockIdx.x * blockDim.x + threadIdx.x;
  if (t >= NE * NH) return;
  int e = t >> 2, hd = t & 3;
  int s = ei[e], d = ei[NE + e];
  float ew[EDIM];
  const float4* ewp = (const float4*)(EW + (size_t)e * EDIM);
  float4 w0 = ewp[0], w1 = ewp[1];
  ew[0] = w0.x; ew[1] = w0.y; ew[2] = w0.z; ew[3] = w0.w;
  ew[4] = w1.x; ew[5] = w1.y; ew[6] = w1.z; ew[7] = w1.w;
  const float* xl = XL + (size_t)s * HC + hd * CC;
  const float* xr = XR + (size_t)d * HC + hd * CC;
  const float* at = att + hd * CC;
  const float* we = We + hd * CC;
  float logit = 0.f;
#pragma unroll 8
  for (int c = 0; c < CC; ++c) {
    float ea = 0.f;
#pragma unroll
    for (int k = 0; k < EDIM; ++k) ea += ew[k] * we[k * HC + c];
    float v = xl[c] + xr[c] + ea;
    v = v > 0.f ? v : 0.2f * v;
    logit += v * at[c];
  }
  LG[t] = logit;
  atomicMax(&MX[d * NH + hd], mapf(logit));
}

// thread per (edge, head): ex = exp(logit - max), atomic segment-sum
__global__ void edge_exp_kernel(const int* __restrict__ ei,
                                float* __restrict__ LG,
                                const unsigned* __restrict__ MX,
                                float* __restrict__ DN) {
  int t = blockIdx.x * blockDim.x + threadIdx.x;
  if (t >= NE * NH) return;
  int e = t >> 2, hd = t & 3;
  int d = ei[NE + e];
  float m  = unmapf(MX[d * NH + hd]);
  float ex = __expf(LG[t] - m);
  LG[t] = ex;
  atomicAdd(&DN[d * NH + hd], ex);
}

// thread per (edge, head): scatter alpha * xl[src] into out[dst]
__global__ void edge_aggr_kernel(const int* __restrict__ ei,
                                 const float* __restrict__ XL,
                                 const float* __restrict__ LG,
                                 const float* __restrict__ DN,
                                 float* __restrict__ OUTA) {
  int t = blockIdx.x * blockDim.x + threadIdx.x;
  if (t >= NE * NH) return;
  int e = t >> 2, hd = t & 3;
  int s = ei[e], d = ei[NE + e];
  float alpha = LG[t] / (DN[d * NH + hd] + 1e-16f);
  const float4* xl = (const float4*)(XL + (size_t)s * HC + hd * CC);
  float* ob = OUTA + (size_t)d * HC + hd * CC;
#pragma unroll
  for (int c4 = 0; c4 < CC / 4; ++c4) {
    float4 v = xl[c4];
    atomicAdd(ob + c4 * 4 + 0, v.x * alpha);
    atomicAdd(ob + c4 * 4 + 1, v.y * alpha);
    atomicAdd(ob + c4 * 4 + 2, v.z * alpha);
    atomicAdd(ob + c4 * 4 + 3, v.w * alpha);
  }
}

// h = out + conv_b + res(h)
__global__ void combine_kernel(const float* __restrict__ OUTA,
                               const float* __restrict__ cb,
                               float* __restrict__ H) {
  int t = blockIdx.x * blockDim.x + threadIdx.x;
  if (t >= NN * HC) return;
  int ch = t & (HC - 1);
  H[t] = OUTA[t] + cb[ch] + H[t];
}

// ---------------------------------------------------------------- launch

extern "C" void kernel_launch(void* const* d_in, const int* in_sizes, int n_in,
                              void* d_out, int out_size, void* d_ws, size_t ws_size,
                              hipStream_t stream) {
  const float* x       = (const float*)d_in[0];
  const int*   ei      = (const int*)d_in[1];
  const float* ew      = (const float*)d_in[2];
  const float* ri_w    = (const float*)d_in[3];
  const float* ri_b    = (const float*)d_in[4];
  const float* ri_g    = (const float*)d_in[5];
  const float* ri_beta = (const float*)d_in[6];
  const float* bn_g    = (const float*)d_in[7];
  const float* bn_b    = (const float*)d_in[8];
  const float* ll_w    = (const float*)d_in[9];
  const float* ll_b    = (const float*)d_in[10];
  const float* lr_w    = (const float*)d_in[11];
  const float* lr_b    = (const float*)d_in[12];
  const float* le_w    = (const float*)d_in[13];
  const float* att     = (const float*)d_in[14];
  const float* cv_b    = (const float*)d_in[15];
  const float* ro_w    = (const float*)d_in[16];
  const float* ro_b    = (const float*)d_in[17];

  size_t off = 0;
  auto alloc = [&](size_t bytes) -> void* {
    void* p = (char*)d_ws + off;
    off += (bytes + 255) & ~(size_t)255;
    return p;
  };
  _Float16* WPri   = (_Float16*)alloc((size_t)HC * HC * 2);
  _Float16* WPll0  = (_Float16*)alloc((size_t)HC * HC * 2);
  _Float16* WPll1  = (_Float16*)alloc((size_t)HC * HC * 2);
  _Float16* WPlr0  = (_Float16*)alloc((size_t)HC * HC * 2);
  _Float16* WPlr1  = (_Float16*)alloc((size_t)HC * HC * 2);
  _Float16* WPro   = (_Float16*)alloc((size_t)HC * 64 * 2);
  _Float16* Ah     = (_Float16*)alloc((size_t)NN * HC * 2);   // f16 GEMM input
  float*    T      = (float*)alloc((size_t)NN * HC * 4);      // pre-BN readin
  float*    Hb     = (float*)alloc((size_t)NN * HC * 4);      // h (residual)
  float*    XL     = (float*)alloc((size_t)NN * HC * 4);
  float*    XR     = (float*)alloc((size_t)NN * HC * 4);
  float*    OUTA   = (float*)alloc((size_t)NN * HC * 4);
  float*    LG     = (float*)alloc((size_t)NE * NH * 4);
  unsigned* MX     = (unsigned*)alloc((size_t)NN * NH * 4);
  float*    DN     = (float*)alloc((size_t)NN * NH * 4);
  float*    STATS  = (float*)alloc(512 * 4);  // SUM|SSQ|MEAN|RS
  float* SUM = STATS, *SSQ = STATS + 128, *MEAN = STATS + 256, *RS = STATS + 384;
  (void)ws_size; (void)in_sizes; (void)n_in; (void)out_size;

  const _Float16* WPll[2] = {WPll0, WPll1};
  const _Float16* WPlr[2] = {WPlr0, WPlr1};

  auto gs = [](int n) { return dim3((unsigned)((n + 255) / 256)); };
  const dim3 b256(256);
  const int rowTiles = NN / 16;  // 3125, exact

  // --- pack all dense weights into WMMA B-operand tiles
  pack_w_kernel<<<gs(HC * HC), b256, 0, stream>>>(ri_w, WPri, 8);
  pack_w_kernel<<<gs(HC * HC), b256, 0, stream>>>(ll_w, WPll0, 8);
  pack_w_kernel<<<gs(HC * HC), b256, 0, stream>>>(ll_w + HC * HC, WPll1, 8);
  pack_w_kernel<<<gs(HC * HC), b256, 0, stream>>>(lr_w, WPlr0, 8);
  pack_w_kernel<<<gs(HC * HC), b256, 0, stream>>>(lr_w + HC * HC, WPlr1, 8);
  pack_w_kernel<<<gs(HC * 64), b256, 0, stream>>>(ro_w, WPro, 4);

  // --- readin: Linear -> BN -> LeakyReLU
  f32_to_f16_kernel<<<gs(NN * HC), b256, 0, stream>>>(x, Ah, NN * HC);
  wmma_gemm_kernel<<<rowTiles, 256, 0, stream>>>(Ah, WPri, ri_b, T, 8);
  zero_f32_kernel<<<1, 256, 0, stream>>>(SUM, 256);
  bn_stats_kernel<<<512, 128, 0, stream>>>(T, SUM, SSQ);
  bn_finalize_kernel<<<1, 128, 0, stream>>>(SUM, SSQ, MEAN, RS);
  bn_act_f32_kernel<<<gs(NN * HC), b256, 0, stream>>>(T, MEAN, RS, ri_g, ri_beta, Hb);

  // --- GATv2 layers (res+)
  for (int l = 0; l < 2; ++l) {
    zero_f32_kernel<<<1, 256, 0, stream>>>(SUM, 256);
    bn_stats_kernel<<<512, 128, 0, stream>>>(Hb, SUM, SSQ);
    bn_finalize_kernel<<<1, 128, 0, stream>>>(SUM, SSQ, MEAN, RS);
    bn_act_f16_kernel<<<gs(NN * HC), b256, 0, stream>>>(Hb, MEAN, RS,
                                                        bn_g + l * HC, bn_b + l * HC, Ah);
    wmma_gemm_kernel<<<rowTiles, 256, 0, stream>>>(Ah, WPll[l], ll_b + l * HC, XL, 8);
    wmma_gemm_kernel<<<rowTiles, 256, 0, stream>>>(Ah, WPlr[l], lr_b + l * HC, XR, 8);

    zero_f32_kernel<<<gs(NN * NH), b256, 0, stream>>>((float*)MX, NN * NH);
    zero_f32_kernel<<<gs(NN * NH), b256, 0, stream>>>(DN, NN * NH);
    zero_f32_kernel<<<gs(NN * HC), b256, 0, stream>>>(OUTA, NN * HC);

    edge_logits_kernel<<<gs(NE * NH), b256, 0, stream>>>(
        ei, XL, XR, ew, le_w + l * EDIM * HC, att + l * HC, LG, MX);
    edge_exp_kernel<<<gs(NE * NH), b256, 0, stream>>>(ei, LG, MX, DN);
    edge_aggr_kernel<<<gs(NE * NH), b256, 0, stream>>>(ei, XL, LG, DN, OUTA);
    combine_kernel<<<gs(NN * HC), b256, 0, stream>>>(OUTA, cv_b + l * HC, Hb);
  }

  // --- readout Linear
  f32_to_f16_kernel<<<gs(NN * HC), b256, 0, stream>>>(Hb, Ah, NN * HC);
  wmma_gemm_kernel<<<rowTiles, 128, 0, stream>>>(Ah, WPro, ro_b, (float*)d_out, 4);
}
